// hyperedge_loss_60026462929137
// MI455X (gfx1250) — matrix operand order repaired and checked by reading, not compiled, not run on hardware
//
#include <hip/hip_runtime.h>
#include <hip/hip_bf16.h>
#include <math.h>

// CDNA5 / gfx1250, wave32. bf16 WMMA 16x16x32 with f32 accumulate.
typedef __attribute__((ext_vector_type(16))) __bf16 v16bf;
typedef __attribute__((ext_vector_type(8)))  __bf16 v8bf;
typedef __attribute__((ext_vector_type(8)))  float  v8f;

constexpr int EDIM = 128;   // D
constexpr int NNEG = 256;   // S
constexpr float TINV = 10.0f;  // 1/temperature

__device__ __forceinline__ float wave_sum32(float x) {
  x += __shfl_xor(x, 1);
  x += __shfl_xor(x, 2);
  x += __shfl_xor(x, 4);
  x += __shfl_xor(x, 8);
  x += __shfl_xor(x, 16);
  return x;
}

// ---------------- kernel 1: per-row L2 norm ----------------
__global__ void norm_kernel(const float* __restrict__ E, float* __restrict__ nrm, int N) {
  int wave = (int)((blockIdx.x * blockDim.x + threadIdx.x) >> 5);
  int lane = threadIdx.x & 31;
  if (wave >= N) return;
  const float4* row = (const float4*)(E + (size_t)wave * EDIM);
  float4 v = row[lane];
  float s = v.x * v.x + v.y * v.y + v.z * v.z + v.w * v.w;
  s = wave_sum32(s);
  if (lane == 0) nrm[wave] = sqrtf(s);
}

// ---------------- kernel 2: gather + normalize negatives, split to bf16 hi/lo ----------------
__global__ void gather_b_kernel(const float* __restrict__ E, const int* __restrict__ negidx,
                                const float* __restrict__ nrm,
                                __bf16* __restrict__ Bhi, __bf16* __restrict__ Blo) {
  int col = blockIdx.x;    // 0..NNEG-1
  int d   = threadIdx.x;   // 0..EDIM-1
  int idx = negidx[col];
  float rn = 1.0f / fmaxf(nrm[idx], 1e-8f);
  float x = E[(size_t)idx * EDIM + d] * rn;
  __bf16 h = (__bf16)x;
  float lo = x - (float)h;
  Bhi[col * EDIM + d] = h;
  Blo[col * EDIM + d] = (__bf16)lo;
}

// ---------------- kernel 3: main. One wave = 16 rows. ----------------
// C tiles: 16 N-tiles of v8f accumulators; bf16x2 split -> 3 WMMA per (kb, tile).
__global__ void __launch_bounds__(256)
nce_main_kernel(const float* __restrict__ E, const int* __restrict__ posidx,
                const float* __restrict__ nrm, const __bf16* __restrict__ Bg, // Bhi||Blo
                float* __restrict__ part, int ntiles) {
  __shared__ __align__(16) __bf16 sB[2 * NNEG * EDIM];  // 128 KB (WGP LDS = 320 KB)
  {
    const uint4* src = (const uint4*)Bg;
    uint4* dst = (uint4*)sB;
    const int nvec = (2 * NNEG * EDIM * 2) / 16;  // 8192
    for (int i = threadIdx.x; i < nvec; i += blockDim.x) dst[i] = src[i];
  }
  __syncthreads();

  int wave = threadIdx.x >> 5;
  int tile = blockIdx.x * 8 + wave;
  if (tile >= ntiles) return;          // whole-wave uniform: EXEC stays all-ones for WMMA

  int lane = threadIdx.x & 31;
  int r = lane & 15;                   // row within tile / col within N-tile
  int h = lane >> 4;                   // lane half
  int row = tile * 16 + r;

  const float* Arow = E + (size_t)row * EDIM;
  int pidx = posidx[row];
  const float* Prow = E + (size_t)pidx * EDIM;

  float nr    = nrm[row];
  float rnA   = 1.0f / fmaxf(nr, 1e-8f);    // cosine path (eps 1e-8)
  float rn12r = 1.0f / fmaxf(nr, 1e-12f);   // positive path (eps 1e-12)
  float rn12p = 1.0f / fmaxf(nrm[pidx], 1e-12f);

  const __bf16* sBhi = sB;
  const __bf16* sBlo = sB + NNEG * EDIM;

  v8f C[16] = {};      // 16 N-tiles x 8 VGPRs f32 accumulators
  float pacc = 0.0f;   // positive dot partial (this lane's K elements)

  #pragma unroll
  for (int kb = 0; kb < 4; ++kb) {          // K blocks of 32
    // A fragment: contiguous K = kb*32 + h*16 .. +15 (matches B)
    int abase = kb * 32 + h * 16;
    float4 a0 = *(const float4*)(Arow + abase);
    float4 a1 = *(const float4*)(Arow + abase + 4);
    float4 a2 = *(const float4*)(Arow + abase + 8);
    float4 a3 = *(const float4*)(Arow + abase + 12);
    float4 p0 = *(const float4*)(Prow + abase);
    float4 p1 = *(const float4*)(Prow + abase + 4);
    float4 p2 = *(const float4*)(Prow + abase + 8);
    float4 p3 = *(const float4*)(Prow + abase + 12);
    pacc += a0.x*p0.x + a0.y*p0.y + a0.z*p0.z + a0.w*p0.w;
    pacc += a1.x*p1.x + a1.y*p1.y + a1.z*p1.z + a1.w*p1.w;
    pacc += a2.x*p2.x + a2.y*p2.y + a2.z*p2.z + a2.w*p2.w;
    pacc += a3.x*p3.x + a3.y*p3.y + a3.z*p3.z + a3.w*p3.w;

    float af[16] = { a0.x*rnA, a0.y*rnA, a0.z*rnA, a0.w*rnA,
                     a1.x*rnA, a1.y*rnA, a1.z*rnA, a1.w*rnA,
                     a2.x*rnA, a2.y*rnA, a2.z*rnA, a2.w*rnA,
                     a3.x*rnA, a3.y*rnA, a3.z*rnA, a3.w*rnA };
    v16bf Ahi, Alo;
    #pragma unroll
    for (int i = 0; i < 16; ++i) {
      __bf16 hh = (__bf16)af[i];
      Ahi[i] = hh;
      Alo[i] = (__bf16)(af[i] - (float)hh);
    }

    #pragma unroll
    for (int t = 0; t < 16; ++t) {
      // B fragment (K order chosen to match A): col = t*16+r, contiguous K = kb*32 + h*16 .. +15
      int coff = (t * 16 + r) * EDIM + kb * 32 + h * 16;   // 32B-aligned
      v16bf Bh = *(const v16bf*)(sBhi + coff);
      v16bf Bl = *(const v16bf*)(sBlo + coff);
      C[t] = __builtin_amdgcn_wmma_f32_16x16x32_bf16(false, Ahi, false, Bh, (short)0, C[t], false, false);
      C[t] = __builtin_amdgcn_wmma_f32_16x16x32_bf16(false, Ahi, false, Bl, (short)0, C[t], false, false);
      C[t] = __builtin_amdgcn_wmma_f32_16x16x32_bf16(false, Alo, false, Bh, (short)0, C[t], false, false);
    }
  }

  // positive logit: lane r and lane r+16 hold disjoint halves of row r's dot
  pacc += __shfl_xor(pacc, 16);
  float plogit = pacc * rn12r * rn12p * TINV;

  // negatives: stable one-pass logsumexp with fixed shift 1/T (max possible logit)
  // C layout: VGPR v, lanes0-15 -> (M=v, N=lane); lanes16-31 -> (M=v+8, N=lane-16)
  float lneg = 0.0f;
  #pragma unroll
  for (int v = 0; v < 8; ++v) {
    float es = 0.0f;
    #pragma unroll
    for (int t = 0; t < 16; ++t) es += __expf(C[t][v] * TINV - TINV);
    es += __shfl_xor(es, 1);
    es += __shfl_xor(es, 2);
    es += __shfl_xor(es, 4);
    es += __shfl_xor(es, 8);   // all 16 lanes of each half now hold the row sum
    lneg += __logf(es);
  }

  // per-lane duplication factors: lneg duplicated x16 per half; plogit duplicated x2
  float val = lneg * (1.0f / 16.0f) - plogit * 0.5f;
  val = wave_sum32(val);
  if (lane == 0)
    part[tile] = val + 16.0f * (TINV - logf(256.0f));  // per-row shift & -log(S), 16 rows
}

// ---------------- kernel 4: deterministic final reduction ----------------
__global__ void reduce_kernel(const float* __restrict__ part, float* __restrict__ out,
                              int ntiles, float invN) {
  __shared__ float s[256];
  float acc = 0.0f;
  for (int i = threadIdx.x; i < ntiles; i += 256) acc += part[i];
  s[threadIdx.x] = acc;
  __syncthreads();
  for (int off = 128; off > 0; off >>= 1) {
    if ((int)threadIdx.x < off) s[threadIdx.x] += s[threadIdx.x + off];
    __syncthreads();
  }
  if (threadIdx.x == 0) out[0] = s[0] * invN;
}

extern "C" void kernel_launch(void* const* d_in, const int* in_sizes, int n_in,
                              void* d_out, int out_size, void* d_ws, size_t ws_size,
                              hipStream_t stream) {
  const float* E  = (const float*)d_in[0];
  const int* pos  = (const int*)d_in[1];
  const int* neg  = (const int*)d_in[2];
  int N = in_sizes[1];                    // 500000
  float* out = (float*)d_out;

  char* ws = (char*)d_ws;
  float*  nrm  = (float*)ws;                                        // N floats
  size_t  offB = (((size_t)N * 4) + 255) & ~(size_t)255;
  __bf16* Bhi  = (__bf16*)(ws + offB);                              // S*D bf16
  __bf16* Blo  = Bhi + (size_t)NNEG * EDIM;                         // S*D bf16 (contiguous)
  float*  part = (float*)(ws + offB + (size_t)2 * NNEG * EDIM * 2); // ntiles floats

  int ntiles = (N + 15) / 16;

  norm_kernel<<<(N + 7) / 8, 256, 0, stream>>>(E, nrm, N);
  gather_b_kernel<<<NNEG, EDIM, 0, stream>>>(E, neg, nrm, Bhi, Blo);
  nce_main_kernel<<<(ntiles + 7) / 8, 256, 0, stream>>>(E, pos, nrm, (const __bf16*)Bhi,
                                                        part, ntiles);
  reduce_kernel<<<1, 256, 0, stream>>>(part, out, ntiles, 1.0f / (float)N);
}
